// GlobalAttention_51694226374799
// MI455X (gfx1250) — matrix operand order repaired and checked by reading, not compiled, not run on hardware
//
#include <hip/hip_runtime.h>
#include <stdint.h>

#define RDIM 384
#define SDIM 2048
#define CEDIM 64
#define NTHREADS 256

typedef __attribute__((ext_vector_type(16))) __bf16 bf16x16;
typedef __attribute__((ext_vector_type(8)))  float  f32x8;

// one-time (weights): round-to-nearest-even pack
static __device__ __forceinline__ uint32_t pack2_bf16_rne(float lo, float hi) {
  uint32_t ua = __float_as_uint(lo);
  uint32_t ub = __float_as_uint(hi);
  ua += 0x7FFFu + ((ua >> 16) & 1u);
  ub += 0x7FFFu + ((ub >> 16) & 1u);
  return (ua >> 16) | (ub & 0xFFFF0000u);
}

// hot path: single v_perm_b32 truncating pack {hi.bf16, lo.bf16}
static __device__ __forceinline__ uint32_t pack2_bf16(float lo, float hi) {
  return __builtin_amdgcn_perm(__float_as_uint(hi), __float_as_uint(lo), 0x07060302u);
}

static __device__ __forceinline__ uint16_t f32_bf16(float x) {
  return (uint16_t)(__float_as_uint(x) >> 16);
}

static __device__ __forceinline__ bf16x16 as_bf16x16(const uint32_t* p) {
  bf16x16 r;
  __builtin_memcpy(&r, p, sizeof(r));
  return r;
}

static __device__ __forceinline__ f32x8 wmma_bf16(const uint32_t* a8, const uint32_t* b8, f32x8 c) {
  return __builtin_amdgcn_wmma_f32_16x16x32_bf16(
      /*neg_a=*/false, as_bf16x16(a8),
      /*neg_b=*/false, as_bf16x16(b8),
      /*c_mod=*/(short)0, c,
      /*reuse_a=*/false, /*reuse_b=*/false);
}

__global__ __launch_bounds__(NTHREADS)
void ga_kernel(const float* __restrict__ m, const float* __restrict__ mask,
               const float* __restrict__ Wq, const float* __restrict__ Wk,
               const float* __restrict__ Wv, const float* __restrict__ Wg,
               const float* __restrict__ bgp, const float* __restrict__ Wo,
               const float* __restrict__ bop, const float* __restrict__ add_to,
               float* __restrict__ out)
{
  __shared__ float    kv_lds[SDIM * 16];   // 128 KB : cols 0-7 = k, 8-15 = v (f32)
  __shared__ uint32_t stage[128 * 32];     // 16 KB  : bf16 staging (m chunk / gated acts)
  __shared__ float    mask_lds[SDIM];      // 8 KB
  __shared__ float    red[8 * 64];         // 2 KB   : per-wave q partials
  __shared__ float    qm[CEDIM];
  __shared__ float    qf[CEDIM];
  __shared__ float    of[CEDIM];
  __shared__ float    den_s;

  const int tid  = threadIdx.x;
  const int lane = tid & 31;
  const int wave = tid >> 5;                 // 0..7
  const int r    = blockIdx.x;

  const float*    mrow    = m + (size_t)r * SDIM * CEDIM;
  uint16_t*       stage_h = (uint16_t*)stage;

  if (tid == 0) den_s = 1e-10f;
  {
    const float4* mk4 = (const float4*)(mask + r * SDIM);
    for (int s4 = tid; s4 < SDIM / 4; s4 += NTHREADS) {
      float4 v = mk4[s4];
      ((float4*)mask_lds)[s4] = v;
    }
  }

  // ---- B fragments for fused [Wk;Wv] -> [K=64, N=16] bf16 (2 K-steps) ----
  uint32_t bkv[2][8];
  {
    const int n     = lane & 15;
    const int khalf = lane >> 4;
    const float* wsrc = (n < 8) ? (Wk + n * CEDIM) : (Wv + (n - 8) * CEDIM);
    #pragma unroll
    for (int f = 0; f < 2; ++f)
      #pragma unroll
      for (int j = 0; j < 8; ++j) {
        int k = f * 32 + khalf * 8 + 2 * (j & 3) + ((j & 4) ? 16 : 0);
        bkv[f][j] = pack2_bf16_rne(wsrc[k], wsrc[k + 1]);
      }
  }
  __syncthreads();

  const int quad = tid & 15;       // channel group: c0 = quad*4
  const int rowg = tid >> 4;       // 0..15
  const int arow = lane & 15;      // A-frag row within 16-row subtile
  const int kadd = (lane >> 4) * 8;// A-frag K offset for lane group
  const int drow = (lane >> 4) * 8;// D-frag row offset
  const int dcol = lane & 15;      // D-frag column

  // ============ Phase A: stream m (b128) -> q partials + k,v (WMMA) ============
  float qp0 = 0.f, qp1 = 0.f, qp2 = 0.f, qp3 = 0.f;
  for (int chunk = 0; chunk < SDIM; chunk += 128) {
    #pragma unroll
    for (int i = 0; i < 8; ++i) {
      int sl = rowg + 16 * i;                     // 0..127
      int s  = chunk + sl;
      float4 v = *(const float4*)(mrow + (size_t)s * CEDIM + quad * 4);
      float msk = mask_lds[s];
      qp0 += v.x * msk; qp1 += v.y * msk; qp2 += v.z * msk; qp3 += v.w * msk;
      stage[sl * 32 + quad * 2 + 0] = pack2_bf16(v.x, v.y);
      stage[sl * 32 + quad * 2 + 1] = pack2_bf16(v.z, v.w);
    }
    __syncthreads();
    {
      const int rbase = wave * 16;
      uint32_t am[2][8];
      #pragma unroll
      for (int f = 0; f < 2; ++f)
        #pragma unroll
        for (int j = 0; j < 8; ++j) {
          int K = f * 32 + 2 * (j & 3) + ((j & 4) ? 16 : 0) + kadd;
          am[f][j] = stage[(rbase + arow) * 32 + (K >> 1)];
        }
      f32x8 acc = {};
      acc = wmma_bf16(am[0], bkv[0], acc);
      acc = wmma_bf16(am[1], bkv[1], acc);
      #pragma unroll
      for (int i = 0; i < 8; ++i)
        kv_lds[(chunk + rbase + drow + i) * 16 + dcol] = acc[i];
    }
    __syncthreads();
  }

  // ---- q = (masked-mean(m) @ Wq^T) * 1/sqrt(C) ----
  qp0 += __shfl_xor(qp0, 16, 32);
  qp1 += __shfl_xor(qp1, 16, 32);
  qp2 += __shfl_xor(qp2, 16, 32);
  qp3 += __shfl_xor(qp3, 16, 32);
  if (lane < 16) {
    red[wave * 64 + lane * 4 + 0] = qp0;
    red[wave * 64 + lane * 4 + 1] = qp1;
    red[wave * 64 + lane * 4 + 2] = qp2;
    red[wave * 64 + lane * 4 + 3] = qp3;
  }
  float dpart = 0.0f;
  for (int s = tid; s < SDIM; s += NTHREADS) dpart += mask_lds[s];
  #pragma unroll
  for (int off = 16; off >= 1; off >>= 1) dpart += __shfl_xor(dpart, off, 32);
  if (lane == 0) atomicAdd(&den_s, dpart);
  __syncthreads();

  if (tid < CEDIM) {
    float qn = 0.0f;
    #pragma unroll
    for (int w = 0; w < 8; ++w) qn += red[w * 64 + tid];
    qm[tid] = qn / den_s;
  }
  __syncthreads();

  if (tid < CEDIM) {
    float acc = 0.0f;
    #pragma unroll 8
    for (int c = 0; c < CEDIM; ++c) acc += qm[c] * Wq[tid * CEDIM + c];
    qf[tid] = acc * 0.35355339059327373f;   // sqrt(1/8)
  }
  __syncthreads();

  // ====== Phase B: online softmax attention, one wave per head ======
  {
    const int h = wave;
    float qh[8];
    #pragma unroll
    for (int c = 0; c < 8; ++c) qh[c] = qf[h * 8 + c];
    float mrun = -3.0e38f, lrun = 0.0f;
    float oacc[8];
    #pragma unroll
    for (int c = 0; c < 8; ++c) oacc[c] = 0.0f;

    for (int s = lane; s < SDIM; s += 32) {
      const float* kvp = &kv_lds[s * 16];
      float x = (mask_lds[s] - 1.0f) * 1e9f;
      #pragma unroll
      for (int c = 0; c < 8; ++c) x += qh[c] * kvp[c];
      float mnew = fmaxf(mrun, x);
      float corr = __expf(mrun - mnew);
      float p    = __expf(x - mnew);
      lrun = lrun * corr + p;
      #pragma unroll
      for (int c = 0; c < 8; ++c) oacc[c] = oacc[c] * corr + p * kvp[8 + c];
      mrun = mnew;
    }
    float mall = mrun;
    #pragma unroll
    for (int off = 16; off >= 1; off >>= 1) mall = fmaxf(mall, __shfl_xor(mall, off, 32));
    float cr = __expf(mrun - mall);
    lrun *= cr;
    #pragma unroll
    for (int c = 0; c < 8; ++c) oacc[c] *= cr;
    #pragma unroll
    for (int off = 16; off >= 1; off >>= 1) {
      lrun += __shfl_xor(lrun, off, 32);
      #pragma unroll
      for (int c = 0; c < 8; ++c) oacc[c] += __shfl_xor(oacc[c], off, 32);
    }
    float inv = 1.0f / lrun;
    if (lane == 0) {
      #pragma unroll
      for (int c = 0; c < 8; ++c) of[h * 8 + c] = oacc[c] * inv;
    }
  }
  __syncthreads();

  // ====== Phase C: gate (WMMA) -> sigmoid*o -> out proj (WMMA) + residual ======
  uint32_t bgf[4][2][8];
  uint32_t bof[4][2][8];
  float bgv[4], bov[4], ofac[4];
  {
    const int n     = lane & 15;
    const int khalf = lane >> 4;
    #pragma unroll
    for (int nt = 0; nt < 4; ++nt) {
      const int nn = nt * 16 + n;
      #pragma unroll
      for (int kt = 0; kt < 2; ++kt)
        #pragma unroll
        for (int j = 0; j < 8; ++j) {
          int k = kt * 32 + khalf * 8 + 2 * (j & 3) + ((j & 4) ? 16 : 0);
          bgf[nt][kt][j] = pack2_bf16_rne(Wg[nn * CEDIM + k], Wg[nn * CEDIM + k + 1]);
          bof[nt][kt][j] = pack2_bf16_rne(Wo[nn * CEDIM + k], Wo[nn * CEDIM + k + 1]);
        }
      bgv[nt]  = bgp[nt * 16 + dcol];
      bov[nt]  = bop[nt * 16 + dcol];
      ofac[nt] = of[nt * 16 + dcol];
    }
  }

  for (int chunk = 0; chunk < SDIM; chunk += 128) {
    __syncthreads();
    #pragma unroll
    for (int i = 0; i < 8; ++i) {
      int sl = rowg + 16 * i;
      int s  = chunk + sl;
      float4 v = *(const float4*)(mrow + (size_t)s * CEDIM + quad * 4);
      stage[sl * 32 + quad * 2 + 0] = pack2_bf16(v.x, v.y);
      stage[sl * 32 + quad * 2 + 1] = pack2_bf16(v.z, v.w);
    }
    __syncthreads();

    const int rbase = wave * 16;
    uint32_t am[2][8];
    #pragma unroll
    for (int f = 0; f < 2; ++f)
      #pragma unroll
      for (int j = 0; j < 8; ++j) {
        int K = f * 32 + 2 * (j & 3) + ((j & 4) ? 16 : 0) + kadd;
        am[f][j] = stage[(rbase + arow) * 32 + (K >> 1)];
      }
    f32x8 gacc[4];
    #pragma unroll
    for (int nt = 0; nt < 4; ++nt) {
      f32x8 z = {};
      z = wmma_bf16(am[0], bgf[nt][0], z);
      z = wmma_bf16(am[1], bgf[nt][1], z);
      gacc[nt] = z;
    }
    // gated activation back into this wave's own 16x64 stage region (bf16)
    #pragma unroll
    for (int nt = 0; nt < 4; ++nt) {
      #pragma unroll
      for (int i = 0; i < 8; ++i) {
        float gp = gacc[nt][i] + bgv[nt];
        float g  = 1.0f / (1.0f + __expf(-gp));
        stage_h[(rbase + drow + i) * 64 + nt * 16 + dcol] = f32_bf16(g * ofac[nt]);
      }
    }
    __syncthreads();

    uint32_t ao[2][8];
    #pragma unroll
    for (int f = 0; f < 2; ++f)
      #pragma unroll
      for (int j = 0; j < 8; ++j) {
        int K = f * 32 + 2 * (j & 3) + ((j & 4) ? 16 : 0) + kadd;
        ao[f][j] = stage[(rbase + arow) * 32 + (K >> 1)];
      }
    #pragma unroll
    for (int nt = 0; nt < 4; ++nt) {
      f32x8 y = {};
      y = wmma_bf16(ao[0], bof[nt][0], y);
      y = wmma_bf16(ao[1], bof[nt][1], y);
      #pragma unroll
      for (int i = 0; i < 8; ++i) {
        int sg = chunk + rbase + drow + i;
        size_t idx = ((size_t)sg * RDIM + r) * CEDIM + nt * 16 + dcol;
        out[idx] = y[i] + bov[nt] + add_to[idx];
      }
    }
  }
}

extern "C" void kernel_launch(void* const* d_in, const int* in_sizes, int n_in,
                              void* d_out, int out_size, void* d_ws, size_t ws_size,
                              hipStream_t stream) {
  const float* m      = (const float*)d_in[0];
  const float* mask   = (const float*)d_in[1];
  const float* Wq     = (const float*)d_in[2];
  const float* Wk     = (const float*)d_in[3];
  const float* Wv     = (const float*)d_in[4];
  const float* Wg     = (const float*)d_in[5];
  const float* bg     = (const float*)d_in[6];
  const float* Wo     = (const float*)d_in[7];
  const float* bo     = (const float*)d_in[8];
  const float* add_to = (const float*)d_in[9];
  float* out          = (float*)d_out;

  ga_kernel<<<dim3(RDIM), dim3(NTHREADS), 0, stream>>>(
      m, mask, Wq, Wk, Wv, Wg, bg, Wo, bo, add_to, out);
}
